// GATModel_57174604644966
// MI455X (gfx1250) — matrix operand order repaired and checked by reading, not compiled, not run on hardware
//
#include <hip/hip_runtime.h>

typedef _Float16 v16h __attribute__((ext_vector_type(16)));
typedef _Float16 v8h  __attribute__((ext_vector_type(8)));
typedef float    v8f  __attribute__((ext_vector_type(8)));

#define NNODES 2048
#define HEADS  8

// ---------------- workspace layout (bytes) ----------------
#define SZ_BUF   (NNODES * 512 * 4)            // 4 MB, f32 activations
#define OFF_BUFA 0
#define OFF_BUFB (SZ_BUF)
#define OFF_H16  (2 * SZ_BUF)                  // f16 h, stored transposed [Cout][2048]
#define OFF_W16  (OFF_H16 + NNODES * 512 * 2)  // f16 W^T [Cout][Fin]
#define OFF_ES   (OFF_W16 + 512 * 512 * 2)     // f32 [H][2048]
#define OFF_ET   (OFF_ES + HEADS * NNODES * 4) // f32 [H][2048]

// ---------------- W [Fin][Cout] f32 -> Wt [Cout][Fin] f16 ----------------
__global__ void convert_w_kernel(const float* __restrict__ W, _Float16* __restrict__ Wt,
                                 int Fin, int Cout) {
  int idx = blockIdx.x * 256 + threadIdx.x;
  if (idx >= Fin * Cout) return;
  int k = idx / Cout, o = idx % Cout;
  Wt[o * Fin + k] = (_Float16)W[idx];
}

// ---------------- h = x @ W  (WMMA f16->f32), write h^T f16 ----------------
// one 16x16 output tile per wave; 8 waves / block
__global__ void transform_kernel(const float* __restrict__ xin, const _Float16* __restrict__ Wt,
                                 _Float16* __restrict__ h16T, int Fin, int Cout) {
  const int wave = threadIdx.x >> 5;
  const int lane = threadIdx.x & 31;
  const int grp  = lane >> 4;      // 0: lanes 0-15, 1: lanes 16-31
  const int row  = lane & 15;
  const int tile = blockIdx.x * 8 + wave;
  const int tm = tile & 127;       // 128 node tiles
  const int tn = tile >> 7;        // Cout/16 feature tiles
  const int i0 = tm * 16;
  const int c0 = tn * 16;

  const float*    xr = xin + (size_t)(i0 + row) * Fin; // A row M = lane%16
  const _Float16* wr = Wt  + (size_t)(c0 + row) * Fin; // B col N = lane%16

  v8f acc = {};
  for (int k0 = 0; k0 < Fin; k0 += 32) {
    // A fragment: element e -> K = k0 + 16*grp + e
    float4 q0 = *(const float4*)(xr + k0 + 16 * grp);
    float4 q1 = *(const float4*)(xr + k0 + 16 * grp + 4);
    float4 q2 = *(const float4*)(xr + k0 + 16 * grp + 8);
    float4 q3 = *(const float4*)(xr + k0 + 16 * grp + 12);
    float xa[16] = {q0.x, q0.y, q0.z, q0.w, q1.x, q1.y, q1.z, q1.w,
                    q2.x, q2.y, q2.z, q2.w, q3.x, q3.y, q3.z, q3.w};
    v16h afrag;
#pragma unroll
    for (int e = 0; e < 16; ++e) afrag[e] = (_Float16)xa[e];
    // B fragment: element e -> K = k0 + 16*grp + e (contiguous in W^T row)
    v16h bfrag = *(const v16h*)(wr + k0 + 16 * grp);
    acc = __builtin_amdgcn_wmma_f32_16x16x32_f16(false, afrag, false, bfrag,
                                                 (short)0, acc, false, false);
  }
  // D: lane holds col N = row, nodes M = r + 8*grp (8 consecutive nodes)
  v8h hh;
#pragma unroll
  for (int r = 0; r < 8; ++r) hh[r] = (_Float16)acc[r];
  *(v8h*)(h16T + (size_t)(c0 + row) * NNODES + i0 + 8 * grp) = hh;
}

// ---------------- e_s / e_t per (node, head) ----------------
__global__ void scores_kernel(const _Float16* __restrict__ h16T,
                              const float* __restrict__ a_s, const float* __restrict__ a_t,
                              float* __restrict__ es, float* __restrict__ et, int C) {
  int idx = blockIdx.x * 256 + threadIdx.x;   // n + 2048*hd
  int n = idx & (NNODES - 1);
  int hd = idx >> 11;
  if (hd >= HEADS) return;
  float s = 0.0f, t = 0.0f;
  for (int c = 0; c < C; ++c) {
    float hv = (float)h16T[(size_t)(hd * C + c) * NNODES + n];
    s += hv * a_s[hd * C + c];
    t += hv * a_t[hd * C + c];
  }
  es[hd * NNODES + n] = s;
  et[hd * NNODES + n] = t;
}

// ---------------- flash-style masked softmax + alpha@h (WMMA) ----------------
// block = 16-node tile, wave = head. NCT = C/16 accumulator tiles.
template <int NCT>
__global__ void attn_kernel(const _Float16* __restrict__ h16T, const float* __restrict__ adj,
                            const float* __restrict__ es, const float* __restrict__ et,
                            const float* __restrict__ bias, float* __restrict__ xout) {
  constexpr int C    = NCT * 16;
  constexpr int Cout = HEADS * C;
  const int hd   = threadIdx.x >> 5;
  const int lane = threadIdx.x & 31;
  const int grp  = lane >> 4;
  const int row  = lane & 15;
  const int i0   = blockIdx.x * 16;

  const float es_row = es[hd * NNODES + i0 + row];
  const float* arow  = adj + (size_t)(i0 + row) * NNODES;
  const float* etr   = et + hd * NNODES;

  float m_run = -3.0e38f, l_run = 0.0f;
  v8f acc[NCT] = {};

  for (int m0 = 0; m0 < NNODES; m0 += 32) {
    __builtin_prefetch(arow + ((m0 + 512) & (NNODES - 1)), 0, 0);
    const int b1 = m0 + 16 * grp;
    const int b2 = m0 + 16 * grp + 8;
    float4 e0 = *(const float4*)(etr + b1);
    float4 e1 = *(const float4*)(etr + b1 + 4);
    float4 e2 = *(const float4*)(etr + b2);
    float4 e3 = *(const float4*)(etr + b2 + 4);
    float4 a0 = *(const float4*)(arow + b1);
    float4 a1 = *(const float4*)(arow + b1 + 4);
    float4 a2 = *(const float4*)(arow + b2);
    float4 a3 = *(const float4*)(arow + b2 + 4);
    float ev[16] = {e0.x, e0.y, e0.z, e0.w, e1.x, e1.y, e1.z, e1.w,
                    e2.x, e2.y, e2.z, e2.w, e3.x, e3.y, e3.z, e3.w};
    float av[16] = {a0.x, a0.y, a0.z, a0.w, a1.x, a1.y, a1.z, a1.w,
                    a2.x, a2.y, a2.z, a2.w, a3.x, a3.y, a3.z, a3.w};
    float sc[16];
#pragma unroll
    for (int e = 0; e < 16; ++e) {
      float t = es_row + ev[e];
      t = t > 0.0f ? t : 0.2f * t;                 // leaky_relu(., 0.2)
      sc[e] = t - 1.0e9f * (1.0f - av[e]);         // edge mask
    }
    float lmax = sc[0];
#pragma unroll
    for (int e = 1; e < 16; ++e) lmax = fmaxf(lmax, sc[e]);
    lmax = fmaxf(lmax, __shfl_xor(lmax, 16, 32));  // combine both K halves of the row
    const float mnew  = fmaxf(m_run, lmax);
    const float scale = __expf(m_run - mnew);
    float psum = 0.0f;
    v16h pa;
#pragma unroll
    for (int e = 0; e < 16; ++e) {
      float p = __expf(sc[e] - mnew);
      psum += p;
      pa[e] = (_Float16)p;
    }
    psum += __shfl_xor(psum, 16, 32);
    l_run = l_run * scale + psum;
    m_run = mnew;
    // rescale accumulators: C/D row j = r + 8*grp, scale for row j lives in lane j (and j+16)
#pragma unroll
    for (int r = 0; r < 8; ++r) {
      float sr = __shfl(scale, r + 8 * grp, 32);
#pragma unroll
      for (int ct = 0; ct < NCT; ++ct) acc[ct][r] *= sr;
    }
#pragma unroll
    for (int ct = 0; ct < NCT; ++ct) {
      v16h bfrag = *(const v16h*)(h16T + (size_t)(hd * C + ct * 16 + row) * NNODES +
                                  m0 + 16 * grp);
      acc[ct] = __builtin_amdgcn_wmma_f32_16x16x32_f16(false, pa, false, bfrag,
                                                       (short)0, acc[ct], false, false);
    }
  }
  // normalize + bias + relu, write f32 activations [node][Cout]
#pragma unroll
  for (int r = 0; r < 8; ++r) {
    float lr  = __shfl(l_run, r + 8 * grp, 32);
    float inv = 1.0f / lr;
    int node  = i0 + 8 * grp + r;
#pragma unroll
    for (int ct = 0; ct < NCT; ++ct) {
      int colg = hd * C + ct * 16 + row;
      float v = acc[ct][r] * inv + bias[colg];
      xout[(size_t)node * Cout + colg] = v > 0.0f ? v : 0.0f;
    }
  }
}

// ---------------- global sum pool over nodes ----------------
__global__ void pool_kernel(const float* __restrict__ xin, int Cout, float* __restrict__ outp) {
  __shared__ float red[256];
  int col = blockIdx.x;
  float s = 0.0f;
  for (int n = threadIdx.x; n < NNODES; n += 256) s += xin[(size_t)n * Cout + col];
  red[threadIdx.x] = s;
  __syncthreads();
  for (int w = 128; w > 0; w >>= 1) {
    if (threadIdx.x < w) red[threadIdx.x] += red[threadIdx.x + w];
    __syncthreads();
  }
  if (threadIdx.x == 0) outp[col] = red[0];
}

// ---------------- one GAT layer ----------------
static void run_layer(const float* xin, int Fin, int C,
                      const float* W, const float* a_s, const float* a_t, const float* b,
                      const float* adj, float* xout, void* ws, hipStream_t stream) {
  const int Cout = HEADS * C;
  char* wsb = (char*)ws;
  _Float16* h16T = (_Float16*)(wsb + OFF_H16);
  _Float16* wt16 = (_Float16*)(wsb + OFF_W16);
  float* es = (float*)(wsb + OFF_ES);
  float* et = (float*)(wsb + OFF_ET);

  int totalW = Fin * Cout;
  convert_w_kernel<<<(totalW + 255) / 256, 256, 0, stream>>>(W, wt16, Fin, Cout);
  int tiles = 128 * (Cout / 16);                 // always a multiple of 8
  transform_kernel<<<tiles / 8, 256, 0, stream>>>(xin, wt16, h16T, Fin, Cout);
  scores_kernel<<<(NNODES * HEADS) / 256, 256, 0, stream>>>(h16T, a_s, a_t, es, et, C);
  if (C == 64)
    attn_kernel<4><<<NNODES / 16, 256, 0, stream>>>(h16T, adj, es, et, b, xout);
  else
    attn_kernel<2><<<NNODES / 16, 256, 0, stream>>>(h16T, adj, es, et, b, xout);
}

extern "C" void kernel_launch(void* const* d_in, const int* in_sizes, int n_in,
                              void* d_out, int out_size, void* d_ws, size_t ws_size,
                              hipStream_t stream) {
  (void)in_sizes; (void)n_in; (void)out_size; (void)ws_size;
  const float* x   = (const float*)d_in[0];
  const float* adj = (const float*)d_in[1];
  auto P = [&](int i) { return (const float*)d_in[i]; };

  char* wsb = (char*)d_ws;
  float* bufA = (float*)(wsb + OFF_BUFA);
  float* bufB = (float*)(wsb + OFF_BUFB);
  float* out  = (float*)d_out;

  // block 1: 128 -> 512 -> 512 -> 512, sum-pool
  run_layer(x,    128, 64, P(2),  P(3),  P(4),  P(5),  adj, bufA, d_ws, stream);
  run_layer(bufA, 512, 64, P(6),  P(7),  P(8),  P(9),  adj, bufB, d_ws, stream);
  run_layer(bufB, 512, 64, P(10), P(11), P(12), P(13), adj, bufA, d_ws, stream);
  pool_kernel<<<512, 256, 0, stream>>>(bufA, 512, out);

  // block 2: 128 -> 256 -> 256 -> 256, sum-pool
  run_layer(x,    128, 32, P(14), P(15), P(16), P(17), adj, bufB, d_ws, stream);
  run_layer(bufB, 256, 32, P(18), P(19), P(20), P(21), adj, bufA, d_ws, stream);
  run_layer(bufA, 256, 32, P(22), P(23), P(24), P(25), adj, bufB, d_ws, stream);
  pool_kernel<<<256, 256, 0, stream>>>(bufB, 256, out + 512);
}
